// Decoder_ARVAE_27530740367609
// MI455X (gfx1250) — matrix operand order repaired and checked by reading, not compiled, not run on hardware
//
#include <hip/hip_runtime.h>

// ---------------- problem constants ----------------
#define B_      2048
#define REALNL  500
#define NL_     504
#define NZ_     50
#define NC_     21
#define GH_     512
#define LRF_    336
#define PER_B   21168   // 336*63 == 168*126 == 84*252 == 42*504
#define ROWS    32      // batch rows per workgroup (two WMMA M tiles -> B-frag reuse x2)
#define NTHR    512     // 16 waves per workgroup
#define HST     520     // LDS stride (bf16 elems) for h buffer, padded vs bank conflicts
#define XST     72      // LDS stride for x_in buffer

typedef __attribute__((ext_vector_type(16))) __bf16        v16bf;
typedef __attribute__((ext_vector_type(8)))  float         v8f;
typedef __attribute__((ext_vector_type(8)))  unsigned int  v8u;
typedef __attribute__((ext_vector_type(4)))  unsigned int  v4u;

__device__ __forceinline__ unsigned short f2bf(float f) {
  unsigned int u = __builtin_bit_cast(unsigned int, f);
  u += 0x7FFFu + ((u >> 16) & 1u);              // round-to-nearest-even
  return (unsigned short)(u >> 16);
}
__device__ __forceinline__ float bf2f(unsigned short s) {
  unsigned int u = ((unsigned int)s) << 16;
  return __builtin_bit_cast(float, u);
}
__device__ __forceinline__ float sigmoidf_(float x) {
  return 1.0f / (1.0f + __expf(-x));
}

__device__ __forceinline__ v16bf make_frag(v4u x0, v4u x1) {
  v8u r;
  r[0] = x0[0]; r[1] = x0[1]; r[2] = x0[2]; r[3] = x0[3];
  r[4] = x1[0]; r[5] = x1[1]; r[6] = x1[2]; r[7] = x1[3];
  return __builtin_bit_cast(v16bf, r);
}

// A fragment (16x32 bf16) from LDS, row-major stride ld (elements).
// ISA 7.12.2: lane<16 -> M=lane, K {kb..kb+7, kb+16..kb+23};
//             lane>=16 -> M=lane-16, K {kb+8..kb+15, kb+24..kb+31}
__device__ __forceinline__ v16bf load_a_lds(const unsigned short* base, int lane, int kb, int ld) {
  int row = lane & 15;
  int off = (lane < 16) ? 0 : 8;
  const unsigned short* p = base + row * ld + kb + off;
  v4u x0 = *(const v4u*)p;          // ds_load_b128
  v4u x1 = *(const v4u*)(p + 16);   // ds_load_b128
  return make_frag(x0, x1);
}

// B fragment (32x16 bf16) from global, weights row-major [n][ld] (= W^T column-major):
// lane<16 -> N=nrow, K {kb..kb+7, kb+16..kb+23}; lane>=16 -> K {kb+8..kb+15, kb+24..kb+31}
__device__ __forceinline__ v16bf load_b_glb(const unsigned short* w, int ld, int nrow, int kb, int lane) {
  const unsigned short* p = w + (size_t)nrow * ld + kb + ((lane < 16) ? 0 : 8);
  v4u x0 = *(const v4u*)p;          // global_load_b128
  v4u x1 = *(const v4u*)(p + 16);   // global_load_b128
  return make_frag(x0, x1);
}

__device__ __forceinline__ v8f wmma_bf16(v16bf a, v16bf b, v8f c) {
  return __builtin_amdgcn_wmma_f32_16x16x32_bf16(false, a, false, b, (short)0, c, false, false);
}

// ---------------- weight conversion kernels ----------------
__global__ void cvt_whh_kernel(const float* __restrict__ src, unsigned short* __restrict__ dst) {
  int i = blockIdx.x * 256 + threadIdx.x;          // 1536*512
  dst[i] = f2bf(src[i]);
}
__global__ void cvt_wih_kernel(const float* __restrict__ src, unsigned short* __restrict__ dst) {
  int i = blockIdx.x * 256 + threadIdx.x;          // 1536*64 (pad K 63->64)
  int n = i >> 6, c = i & 63;
  dst[i] = (c < 63) ? f2bf(src[n * 63 + c]) : (unsigned short)0;
}
__global__ void cvt_wout_kernel(const float* __restrict__ src, unsigned short* __restrict__ dst) {
  int i = blockIdx.x * 256 + threadIdx.x;          // 32*512 (pad N 21->32)
  int n = i >> 9, k = i & 511;
  dst[i] = (n < NC_) ? f2bf(src[n * 512 + k]) : (unsigned short)0;
}

// ---------------- upsampler: dense ----------------
__global__ __launch_bounds__(256)
void dense_kernel(const float* __restrict__ z, const float* __restrict__ W,
                  const float* __restrict__ bias, float* __restrict__ out) {
  size_t gid = (size_t)blockIdx.x * 256 + threadIdx.x;   // B_*PER_B outputs
  int b = (int)(gid / PER_B);
  int r = (int)(gid % PER_B);
  const float* zp = z + (size_t)b * NZ_;
  const float* wp = W + (size_t)r * NZ_;
  float acc = bias[r];
#pragma unroll 10
  for (int k = 0; k < NZ_; ++k) acc += zp[k] * wp[k];
  out[gid] = acc;   // layout [b][c][t], t contiguous (r = c*63+t)
}

// ---------------- upsampler: deconv k=2 s=2 + BN(eval) + PReLU ----------------
__global__ __launch_bounds__(256)
void deconv_bn_prelu_kernel(const float* __restrict__ in, const float* __restrict__ W,
                            const float* __restrict__ g, const float* __restrict__ bb,
                            const float* __restrict__ m, const float* __restrict__ vv,
                            const float* __restrict__ alpha,
                            float* __restrict__ out_f32, unsigned short* __restrict__ out_ups,
                            int cin, int cout, int Tin, int last) {
  size_t gid = (size_t)blockIdx.x * 256 + threadIdx.x;   // B_*PER_B outputs
  int b   = (int)(gid / PER_B);
  int rem = (int)(gid % PER_B);
  int Tout = 2 * Tin;
  int o   = rem / Tout;
  int tt  = rem - o * Tout;
  int tin = tt >> 1;
  int k   = tt & 1;
  const float* ip = in + (size_t)b * PER_B + tin;
  const float* wp = W + (size_t)o * 2 + k;
  float acc = 0.0f;
  for (int c = 0; c < cin; ++c)
    acc += ip[(size_t)c * Tin] * wp[(size_t)c * cout * 2];
  float scale = g[o] * rsqrtf(vv[o] + 1e-5f);
  float val = (acc - m[o]) * scale + bb[o];
  float a = alpha[0];
  val = (val >= 0.0f) ? val : a * val;
  if (!last) {
    out_f32[gid] = val;                                   // [b][o][tt]
  } else {
    out_ups[((size_t)tt * B_ + b) * 42 + o] = f2bf(val);  // [t][b][ch] bf16 for GRU
  }
}

// ---------------- autoregressive GRU decode (persistent per batch slice) ----------------
// 64 blocks x 512 threads (16 waves). Wave w owns hidden channels [w*32, w*32+32)
// (2 N-tiles per gate) across BOTH 16-row M tiles: each B fragment feeds 2 WMMAs.
__global__ __launch_bounds__(NTHR, 1)
void gru_ar_kernel(const unsigned short* __restrict__ ups,   // [NL][B][42] bf16
                   const unsigned short* __restrict__ whh,   // [1536][512] bf16
                   const unsigned short* __restrict__ wih,   // [1536][64]  bf16
                   const unsigned short* __restrict__ wout,  // [32][512]   bf16
                   const float* __restrict__ b_ih, const float* __restrict__ b_hh,
                   const float* __restrict__ w_px, const float* __restrict__ b_px,
                   const float* __restrict__ b_out,
                   float* __restrict__ out)                  // [B][500][21] f32
{
  __shared__ __align__(16) unsigned short hbuf[ROWS * HST];  // gru hidden, bf16
  __shared__ __align__(16) unsigned short xbuf[ROWS * XST];  // x_in, bf16 (64 cols used)
  __shared__ float lgbuf[ROWS * 32];
  __shared__ int   idxbuf[ROWS];

  const int tid   = threadIdx.x;
  const int lane  = tid & 31;
  const int wave  = tid >> 5;           // 16 waves
  const int b0    = blockIdx.x * ROWS;
  const int colw  = lane & 15;
  const int rbase = (lane < 16) ? 0 : 8;

  for (int i = tid; i < ROWS * HST; i += NTHR) hbuf[i] = 0;
  if (tid < ROWS) idxbuf[tid] = -1;     // initial x1h == zeros (not one-hot)
  __syncthreads();

  float brr[2], bzz[2], bni[2], bnh[2];
#pragma unroll
  for (int j = 0; j < 2; ++j) {
    int c = wave * 32 + j * 16 + colw;
    brr[j] = b_ih[c]            + b_hh[c];
    bzz[j] = b_ih[GH_ + c]      + b_hh[GH_ + c];
    bni[j] = b_ih[2 * GH_ + c];
    bnh[j] = b_hh[2 * GH_ + c];
  }

  for (int t = 0; t < REALNL; ++t) {
    // ---- stage x_in = [ups(42) | px(21) | pad0] as bf16 ----
    for (int i = tid; i < ROWS * 64; i += NTHR) {
      int r = i >> 6, c = i & 63;
      unsigned short v;
      if (c < 42) {
        v = ups[((size_t)t * B_ + (b0 + r)) * 42 + c];
      } else if (c < 63) {
        int j = c - 42;
        int idx = idxbuf[r];            // one_hot @ w_px^T == column gather
        float f = b_px[j] + (idx >= 0 ? w_px[j * NC_ + idx] : 0.0f);
        v = f2bf(f);
      } else v = 0;
      xbuf[r * XST + c] = v;
    }
    __syncthreads();

    v8f zero = {};
    v8f accR[2][2], accZ[2][2], accNi[2][2], accNh[2][2];   // [m-tile][n-tile]
#pragma unroll
    for (int m = 0; m < 2; ++m)
#pragma unroll
      for (int j = 0; j < 2; ++j) {
        accR[m][j] = zero; accZ[m][j] = zero; accNi[m][j] = zero; accNh[m][j] = zero;
      }

    // ---- gh = h @ w_hh^T : [32,512]x[512,1536], K-steps of 32 ----
    for (int kk = 0; kk < 16; ++kk) {
      int kb = kk * 32;
      v16bf a0 = load_a_lds(hbuf,             lane, kb, HST);
      v16bf a1 = load_a_lds(hbuf + 16 * HST,  lane, kb, HST);
#pragma unroll
      for (int j = 0; j < 2; ++j) {
        int n0 = wave * 32 + j * 16 + colw;
        v16bf br = load_b_glb(whh, GH_, n0,           kb, lane);
        v16bf bz = load_b_glb(whh, GH_, GH_ + n0,     kb, lane);
        v16bf bn = load_b_glb(whh, GH_, 2 * GH_ + n0, kb, lane);
        accR[0][j]  = wmma_bf16(a0, br, accR[0][j]);
        accR[1][j]  = wmma_bf16(a1, br, accR[1][j]);
        accZ[0][j]  = wmma_bf16(a0, bz, accZ[0][j]);
        accZ[1][j]  = wmma_bf16(a1, bz, accZ[1][j]);
        accNh[0][j] = wmma_bf16(a0, bn, accNh[0][j]);
        accNh[1][j] = wmma_bf16(a1, bn, accNh[1][j]);
      }
    }
    // ---- gi = x_in @ w_ih^T : [32,64]x[64,1536] ----
#pragma unroll
    for (int kk = 0; kk < 2; ++kk) {
      int kb = kk * 32;
      v16bf a0 = load_a_lds(xbuf,            lane, kb, XST);
      v16bf a1 = load_a_lds(xbuf + 16 * XST, lane, kb, XST);
#pragma unroll
      for (int j = 0; j < 2; ++j) {
        int n0 = wave * 32 + j * 16 + colw;
        v16bf br = load_b_glb(wih, 64, n0,           kb, lane);
        v16bf bz = load_b_glb(wih, 64, GH_ + n0,     kb, lane);
        v16bf bn = load_b_glb(wih, 64, 2 * GH_ + n0, kb, lane);
        accR[0][j]  = wmma_bf16(a0, br, accR[0][j]);
        accR[1][j]  = wmma_bf16(a1, br, accR[1][j]);
        accZ[0][j]  = wmma_bf16(a0, bz, accZ[0][j]);
        accZ[1][j]  = wmma_bf16(a1, bz, accZ[1][j]);
        accNi[0][j] = wmma_bf16(a0, bn, accNi[0][j]);
        accNi[1][j] = wmma_bf16(a1, bn, accNi[1][j]);
      }
    }
    __syncthreads();   // all waves done reading old h from LDS

    // ---- gates + hidden update (lane owns col c; rows m*16 + rbase..rbase+7) ----
#pragma unroll
    for (int m = 0; m < 2; ++m) {
#pragma unroll
      for (int j = 0; j < 2; ++j) {
        int c = wave * 32 + j * 16 + colw;
#pragma unroll
        for (int rr = 0; rr < 8; ++rr) {
          int row = m * 16 + rbase + rr;
          float r_ = sigmoidf_(accR[m][j][rr] + brr[j]);
          float z_ = sigmoidf_(accZ[m][j][rr] + bzz[j]);
          float n_ = tanhf(accNi[m][j][rr] + bni[j] + r_ * (accNh[m][j][rr] + bnh[j]));
          float hold = bf2f(hbuf[row * HST + c]);
          float hn = (1.0f - z_) * n_ + z_ * hold;
          hbuf[row * HST + c] = f2bf(hn);
        }
      }
    }
    __syncthreads();   // h_new visible to all waves

    // ---- logits = h_new @ w_out^T (+b_out): 2 M-tiles x 2 padded col-tiles, waves 0..3 ----
    if (wave < 4) {
      const int m  = wave >> 1;
      const int ct = wave & 1;
      v8f accL = zero;
      for (int kk = 0; kk < 16; ++kk) {
        v16bf a = load_a_lds(hbuf + m * 16 * HST, lane, kk * 32, HST);
        accL = wmma_bf16(a, load_b_glb(wout, GH_, ct * 16 + colw, kk * 32, lane), accL);
      }
      int c = ct * 16 + colw;
      float bias = (c < NC_) ? b_out[c] : -1e30f;   // pad cols never win argmax
#pragma unroll
      for (int rr = 0; rr < 8; ++rr)
        lgbuf[(m * 16 + rbase + rr) * 32 + c] = accL[rr] + bias;
    }
    __syncthreads();

    // ---- argmax (first-max, as jnp.argmax) + emit logits ----
    if (tid < ROWS) {
      int best = 0; float bv = lgbuf[tid * 32];
      for (int c = 1; c < NC_; ++c) {
        float v = lgbuf[tid * 32 + c];
        if (v > bv) { bv = v; best = c; }
      }
      idxbuf[tid] = best;
    }
    for (int i = tid; i < ROWS * NC_; i += NTHR) {
      int r = i / NC_, c = i - r * NC_;
      out[((size_t)(b0 + r) * REALNL + t) * NC_ + c] = lgbuf[r * 32 + c];
    }
    __syncthreads();
  }
}

// ---------------- launcher ----------------
extern "C" void kernel_launch(void* const* d_in, const int* in_sizes, int n_in,
                              void* d_out, int out_size, void* d_ws, size_t ws_size,
                              hipStream_t stream) {
  (void)in_sizes; (void)n_in; (void)out_size; (void)ws_size;
  const float* z       = (const float*)d_in[1];
  const float* dense_W = (const float*)d_in[3];
  const float* dense_b = (const float*)d_in[4];
  const float* dcW[3]  = {(const float*)d_in[5],  (const float*)d_in[11], (const float*)d_in[17]};
  const float* bng[3]  = {(const float*)d_in[6],  (const float*)d_in[12], (const float*)d_in[18]};
  const float* bnb[3]  = {(const float*)d_in[7],  (const float*)d_in[13], (const float*)d_in[19]};
  const float* bnm[3]  = {(const float*)d_in[8],  (const float*)d_in[14], (const float*)d_in[20]};
  const float* bnv[3]  = {(const float*)d_in[9],  (const float*)d_in[15], (const float*)d_in[21]};
  const float* pre[3]  = {(const float*)d_in[10], (const float*)d_in[16], (const float*)d_in[22]};
  const float* w_px  = (const float*)d_in[23];
  const float* b_px  = (const float*)d_in[24];
  const float* w_ih  = (const float*)d_in[25];
  const float* w_hh  = (const float*)d_in[26];
  const float* b_ih  = (const float*)d_in[27];
  const float* b_hh  = (const float*)d_in[28];
  const float* w_out = (const float*)d_in[29];
  const float* b_out = (const float*)d_in[30];

  char* ws = (char*)d_ws;
  unsigned short* whh_bf  = (unsigned short*)(ws);                      // 1536*512*2  = 1,572,864 B
  unsigned short* wih_bf  = (unsigned short*)(ws + 1572864);            // 1536*64*2   =   196,608 B
  unsigned short* wout_bf = (unsigned short*)(ws + 1769472);            // 32*512*2    =    32,768 B
  float*          bufA    = (float*)(ws + 1802240);                     // B*PER_B f32 = 173,408,256 B
  float*          bufB    = (float*)(ws + 1802240 + 173408256ull);
  unsigned short* ups     = (unsigned short*)(ws + 1802240 + 2ull * 173408256ull); // 504*2048*42*2

  cvt_whh_kernel<<<(1536 * 512) / 256, 256, 0, stream>>>(w_hh, whh_bf);
  cvt_wih_kernel<<<(1536 * 64) / 256, 256, 0, stream>>>(w_ih, wih_bf);
  cvt_wout_kernel<<<(32 * 512) / 256, 256, 0, stream>>>(w_out, wout_bf);

  const int blocksUP = (B_ * PER_B) / 256;  // 169,344
  dense_kernel<<<blocksUP, 256, 0, stream>>>(z, dense_W, dense_b, bufA);
  deconv_bn_prelu_kernel<<<blocksUP, 256, 0, stream>>>(bufA, dcW[0], bng[0], bnb[0], bnm[0], bnv[0], pre[0],
                                                       bufB, nullptr, 336, 168, 63, 0);
  deconv_bn_prelu_kernel<<<blocksUP, 256, 0, stream>>>(bufB, dcW[1], bng[1], bnb[1], bnm[1], bnv[1], pre[1],
                                                       bufA, nullptr, 168, 84, 126, 0);
  deconv_bn_prelu_kernel<<<blocksUP, 256, 0, stream>>>(bufA, dcW[2], bng[2], bnb[2], bnm[2], bnv[2], pre[2],
                                                       nullptr, ups, 84, 42, 252, 1);

  gru_ar_kernel<<<B_ / ROWS, NTHR, 0, stream>>>(ups, whh_bf, wih_bf, wout_bf,
                                                b_ih, b_hh, w_px, b_px, b_out, (float*)d_out);
}
